// HiResPrecipNet_superres_89223650607427
// MI455X (gfx1250) — matrix operand order repaired and checked by reading, compiled, not run on hardware
//
#include <hip/hip_runtime.h>
#include <math.h>

// ---------------- problem constants ----------------
#define N_LOW   8000
#define N_HIGH  50000
#define N_L25   (N_LOW * 25)         /* 200000 */
#define E_LL    32000
#define E_LH    (N_HIGH * 25)        /* 1250000 */
#define E_HH    (N_HIGH * 16)        /* 800000  */
#define BN_EPS  1e-5f

typedef _Float16 half_t;
typedef __attribute__((ext_vector_type(8)))  _Float16 v8h;
typedef __attribute__((ext_vector_type(16))) _Float16 v16h;
typedef __attribute__((ext_vector_type(8)))  float    v8f;

// ---------------- small device helpers ----------------
__device__ __forceinline__ float lrelu02(float x) { return x >= 0.f ? x : 0.2f * x; }

__device__ __forceinline__ void atomicMaxF(float* addr, float val) {
  unsigned int* ua = (unsigned int*)addr;
  unsigned int old = *ua;
  while (__uint_as_float(old) < val) {
    unsigned int assumed = old;
    old = atomicCAS(ua, assumed, __float_as_uint(val));
    if (old == assumed) break;
  }
}

// ---------------- WMMA GEMM on pre-converted f16 operands ----------------
// A16: [M, KP] f16 row-major (zero-padded cols), M multiple of 16.
// Wt16: [Np, KP] f16 (W transposed, zero-padded), Np multiple of 16.
// C[M,N] f32 (+bias, +ReLU). One wave per 16x16 tile; branch-free inner loop,
// fragment loads are contiguous 16B vectors matching the WMMA layouts.
template<int KP>
__global__ __launch_bounds__(32)
void k_wmma_gemm16(const half_t* __restrict__ A,
                   const half_t* __restrict__ Wt,
                   const float* __restrict__ bias,
                   float* __restrict__ C, int ldc,
                   int N, int relu)
{
  const int tm   = blockIdx.x * 16;
  const int tn   = blockIdx.y * 16;
  const int lane = (int)threadIdx.x;
  const int hs   = lane >> 4;
  const int lr   = lane & 15;

  const half_t* arow = A  + (size_t)(tm + lr) * KP + hs * 8;   // A frag: K = k0+hs*8+{0..7}, +16
  const half_t* brow = Wt + (size_t)(tn + lr) * KP + hs * 8;   // B frag: K = k0+hs*8+{0..7}, +16

  v8f acc = {};
#pragma unroll
  for (int k0 = 0; k0 < KP; k0 += 32) {
    v8h a0 = *(const v8h*)(arow + k0);
    v8h a1 = *(const v8h*)(arow + k0 + 16);
    v8h b0 = *(const v8h*)(brow + k0);
    v8h b1 = *(const v8h*)(brow + k0 + 16);
    v16h af = __builtin_shufflevector(a0, a1, 0,1,2,3,4,5,6,7,8,9,10,11,12,13,14,15);
    v16h bf = __builtin_shufflevector(b0, b1, 0,1,2,3,4,5,6,7,8,9,10,11,12,13,14,15);
    acc = __builtin_amdgcn_wmma_f32_16x16x32_f16(false, af, false, bf,
                                                 (short)0, acc, false, false);
  }
  const int n = tn + lr;
  if (n < N) {
    float bv = bias ? bias[n] : 0.f;
#pragma unroll
    for (int i = 0; i < 8; ++i) {
      int row = tm + i + (hs ? 8 : 0);
      float v = acc[i] + bv;
      if (relu) v = fmaxf(v, 0.f);
      C[(size_t)row * (size_t)ldc + n] = v;
    }
  }
}

// ---------------- conversion kernels (coalesced) ----------------
// f32 [M,K] (lda) -> f16 [M,Kp] zero-padded
__global__ void k_cvt_a16(const float* __restrict__ A, int lda, int M, int K, int Kp,
                          half_t* __restrict__ out) {
  int i = blockIdx.x * blockDim.x + threadIdx.x;
  if (i < M * Kp) {
    int r = i / Kp, c = i % Kp;
    out[i] = (c < K) ? (half_t)A[(size_t)r * lda + c] : (half_t)0.f;
  }
}

// W f32 [K,N] (ldw) -> Wt f16 [Np,Kp] zero-padded (transposed)
__global__ void k_cvt_wt16(const float* __restrict__ W, int ldw, int K, int N,
                           int Kp, int Np, half_t* __restrict__ out) {
  int i = blockIdx.x * blockDim.x + threadIdx.x;
  if (i < Np * Kp) {
    int n = i / Kp, k = i % Kp;
    out[i] = (n < N && k < K) ? (half_t)W[(size_t)k * ldw + n] : (half_t)0.f;
  }
}

// ---------------- utility kernels ----------------
__global__ void k_fill(float* p, float v, int n) {
  int i = blockIdx.x * blockDim.x + threadIdx.x;
  if (i < n) p[i] = v;
}

__global__ void k_deg(const int* __restrict__ dst, float* deg, int n) {
  int e = blockIdx.x * blockDim.x + threadIdx.x;
  if (e < n) atomicAdd(&deg[dst[e]], 1.f);
}

__global__ void k_rsqrt_deg(float* deg, int n) {
  int i = blockIdx.x * blockDim.x + threadIdx.x;
  if (i < n) deg[i] = rsqrtf(fmaxf(deg[i], 1.f));
}

__global__ void k_gcn_self(const float* __restrict__ h, const float* __restrict__ dis,
                           const float* __restrict__ b, float* __restrict__ out) {
  int i = blockIdx.x * blockDim.x + threadIdx.x;
  if (i < N_LOW * 3125) {
    int n = i / 3125, c = i % 3125;
    float d = dis[n];
    out[i] = h[i] * d * d + b[c];
  }
}

__global__ void k_gcn_edges(const int* __restrict__ src, const int* __restrict__ dst,
                            const float* __restrict__ dis, const float* __restrict__ h,
                            float* __restrict__ out) {
  int e = blockIdx.x;
  int s = src[e], d = dst[e];
  float w = dis[s] * dis[d];
  const float* hp = h + (size_t)s * 3125;
  float* op = out + (size_t)d * 3125;
  for (int c = threadIdx.x; c < 3125; c += blockDim.x)
    atomicAdd(&op[c], hp[c] * w);
}

// pixel shuffle + permute, writing padded f16 A-matrix [200000, 128] directly
__global__ void k_pixshuf16(const float* __restrict__ agg, half_t* __restrict__ enc16) {
  int i = blockIdx.x * blockDim.x + threadIdx.x;
  if (i < N_LOW * 3125) {
    int n = i / 3125, rem = i % 3125;
    int v = rem / 625, r2 = rem % 625;
    int c = r2 / 25, r = r2 % 25;
    enc16[(size_t)(n * 25 + r) * 128 + v * 25 + c] = (half_t)agg[i];
  }
}

__global__ void k_hr_th(const float* __restrict__ z, const float* __restrict__ wr,
                        const float* __restrict__ br, float* __restrict__ hr) {
  int i = blockIdx.x * blockDim.x + threadIdx.x;
  if (i < N_HIGH * 100) {
    int d = i / 100, j = i % 100;
    hr[i] = z[d] * wr[j] + br[j];
  }
}

// ---------------- GATv2 edge passes: wave per (edge, head) ----------------
template<int OC>
__global__ void k_gat_logit_w(const int* __restrict__ src, const int* __restrict__ dst,
                              int E, int nself, int H,
                              const float* __restrict__ hl, const float* __restrict__ hr,
                              const float* __restrict__ att,
                              float* __restrict__ logit, float* __restrict__ m,
                              float* __restrict__ cnt) {
  int wid  = blockIdx.x * (blockDim.x >> 5) + ((int)threadIdx.x >> 5);
  int lane = (int)threadIdx.x & 31;
  int tot  = (E + nself) * H;
  if (wid >= tot) return;
  int e = wid / H, h = wid % H;
  int s, d;
  if (e < E) { s = src[e]; d = dst[e]; } else { s = d = e - E; }
  const int HC = H * OC;
  const float* pl = hl + (size_t)s * HC + h * OC;
  const float* pr = hr + (size_t)d * HC + h * OC;
  const float* pa = att + h * OC;
  float acc = 0.f;
#pragma unroll
  for (int c = lane; c < OC; c += 32) acc += lrelu02(pl[c] + pr[c]) * pa[c];
#pragma unroll
  for (int off = 16; off > 0; off >>= 1) acc += __shfl_xor(acc, off, 32);
  if (lane == 0) {
    logit[wid] = acc;
    atomicMaxF(&m[(size_t)d * H + h], acc);
    if (h == 0) atomicAdd(&cnt[d], 1.f);
  }
}

__global__ void k_gat_exp(const int* __restrict__ dst, int E, int nself, int H,
                          const float* __restrict__ m, float* __restrict__ logit,
                          float* __restrict__ den) {
  int t = blockIdx.x * blockDim.x + threadIdx.x;
  int tot = (E + nself) * H;
  if (t >= tot) return;
  int e = t / H, h = t % H;
  int d = (e < E) ? dst[e] : e - E;
  float a = expf(logit[t] - m[(size_t)d * H + h]);
  logit[t] = a;
  atomicAdd(&den[(size_t)d * H + h], a);
}

template<int OC>
__global__ void k_gat_scatter_w(const int* __restrict__ src, const int* __restrict__ dst,
                                int E, int nself, int H,
                                const float* __restrict__ hl,
                                const float* __restrict__ a, const float* __restrict__ den,
                                float* __restrict__ ssum) {
  int wid  = blockIdx.x * (blockDim.x >> 5) + ((int)threadIdx.x >> 5);
  int lane = (int)threadIdx.x & 31;
  int tot  = (E + nself) * H;
  if (wid >= tot) return;
  int e = wid / H, h = wid % H;
  int s, d;
  if (e < E) { s = src[e]; d = dst[e]; } else { s = d = e - E; }
  const int HC = H * OC;
  float w = a[wid] / fmaxf(den[(size_t)d * H + h], 1e-16f);
  const float* pl = hl + (size_t)s * HC + h * OC;
  float* po = ssum + (size_t)d * HC + h * OC;
#pragma unroll
  for (int c = lane; c < OC; c += 32) atomicAdd(&po[c], w * pl[c]);
}

// th GAT finalize: mean-aggr, mean over 4 heads, +bias; builds x26 concat
__global__ void k_th_final(const float* __restrict__ ssum, const float* __restrict__ cnt,
                           const float* __restrict__ bias, const float* __restrict__ z,
                           float* __restrict__ x26) {
  int i = blockIdx.x * blockDim.x + threadIdx.x;
  if (i < N_HIGH * 25) {
    int d = i / 25, c = i % 25;
    float inv = 1.f / fmaxf(cnt[d], 1.f);
    float v = 0.f;
    for (int h = 0; h < 4; ++h) v += ssum[(size_t)d * 100 + h * 25 + c];
    v = v * 0.25f * inv + bias[c];
    x26[(size_t)d * 26 + 1 + c] = v;
    if (c == 0) x26[(size_t)d * 26] = z[d];
  }
}

// hh GAT finalize: mean-aggr (concat=True), +bias, optional ReLU, f32 out
__global__ void k_hh_final(const float* __restrict__ ssum, const float* __restrict__ cnt,
                           const float* __restrict__ bias, int HC, int relu,
                           float* __restrict__ out) {
  int i = blockIdx.x * blockDim.x + threadIdx.x;
  if (i < N_HIGH * HC) {
    int d = i / HC, j = i % HC;
    float v = ssum[i] / fmaxf(cnt[d], 1.f) + bias[j];
    if (relu) v = fmaxf(v, 0.f);
    out[i] = v;
  }
}

// ---------------- BatchNorm (train-mode batch stats) ----------------
__global__ void k_bn_stats(const float* __restrict__ x, int Nrow, int C,
                           float* __restrict__ stats) {
  int ch = blockIdx.x;
  __shared__ float ss[256], sq[256];
  float s = 0.f, q = 0.f;
  for (int i = threadIdx.x; i < Nrow; i += blockDim.x) {
    float v = x[(size_t)i * C + ch];
    s += v; q += v * v;
  }
  ss[threadIdx.x] = s; sq[threadIdx.x] = q;
  __syncthreads();
  for (int o = 128; o > 0; o >>= 1) {
    if ((int)threadIdx.x < o) { ss[threadIdx.x] += ss[threadIdx.x + o]; sq[threadIdx.x] += sq[threadIdx.x + o]; }
    __syncthreads();
  }
  if (threadIdx.x == 0) {
    float mu  = ss[0] / (float)Nrow;
    float var = sq[0] / (float)Nrow - mu * mu;
    stats[ch * 2]     = mu;
    stats[ch * 2 + 1] = rsqrtf(var + BN_EPS);
  }
}

// BN apply fused with f16 conversion + zero padding (output feeds WMMA GEMMs only)
__global__ void k_bn_apply16(const float* __restrict__ x, const float* __restrict__ stats,
                             const float* __restrict__ g, const float* __restrict__ b,
                             int Nrow, int C, int Kp, int relu, half_t* __restrict__ y) {
  int i = blockIdx.x * blockDim.x + threadIdx.x;
  if (i < Nrow * Kp) {
    int r = i / Kp, c = i % Kp;
    float v = 0.f;
    if (c < C) {
      v = (x[(size_t)r * C + c] - stats[c * 2]) * stats[c * 2 + 1] * g[c] + b[c];
      if (relu) v = fmaxf(v, 0.f);
    }
    y[i] = (half_t)v;
  }
}

// ---------------- host orchestration ----------------
extern "C" void kernel_launch(void* const* d_in, const int* in_sizes, int n_in,
                              void* d_out, int out_size, void* d_ws, size_t ws_size,
                              hipStream_t stream) {
  (void)in_sizes; (void)n_in; (void)out_size;

  const float* x_low  = (const float*)d_in[0];
  const float* z_std  = (const float*)d_in[1];
  const float* gcn_w  = (const float*)d_in[2];
  const float* gcn_b  = (const float*)d_in[3];
  const float* th_wl  = (const float*)d_in[4];
  const float* th_bl  = (const float*)d_in[5];
  const float* th_wr  = (const float*)d_in[6];
  const float* th_br  = (const float*)d_in[7];
  const float* th_att = (const float*)d_in[8];
  const float* th_bias= (const float*)d_in[9];
  const float* bn0_g  = (const float*)d_in[10];
  const float* bn0_b  = (const float*)d_in[11];
  const float* g1_wl  = (const float*)d_in[12];
  const float* g1_bl  = (const float*)d_in[13];
  const float* g1_wr  = (const float*)d_in[14];
  const float* g1_br  = (const float*)d_in[15];
  const float* g1_att = (const float*)d_in[16];
  const float* g1_bias= (const float*)d_in[17];
  const float* bn_g   = (const float*)d_in[18];
  const float* bn_b   = (const float*)d_in[19];
  const float* gm_wl  = (const float*)d_in[20];
  const float* gm_bl  = (const float*)d_in[21];
  const float* gm_wr  = (const float*)d_in[22];
  const float* gm_br  = (const float*)d_in[23];
  const float* gm_att = (const float*)d_in[24];
  const float* gm_bias= (const float*)d_in[25];
  const float* g5_wl  = (const float*)d_in[26];
  const float* g5_bl  = (const float*)d_in[27];
  const float* g5_wr  = (const float*)d_in[28];
  const float* g5_br  = (const float*)d_in[29];
  const float* g5_att = (const float*)d_in[30];
  const float* g5_bias= (const float*)d_in[31];
  const float* p_w1   = (const float*)d_in[32];
  const float* p_b1   = (const float*)d_in[33];
  const float* p_w2   = (const float*)d_in[34];
  const float* p_b2   = (const float*)d_in[35];
  const float* p_w3   = (const float*)d_in[36];
  const float* p_b3   = (const float*)d_in[37];
  const int*   ei_ll  = (const int*)d_in[38];
  const int*   src_l2h= (const int*)d_in[39];
  const int*   dst_l2h= (const int*)d_in[40];
  const int*   ei_hh  = (const int*)d_in[41];

  // ---- workspace arena (float words) ----
  float* w = (float*)d_ws;
  size_t off = 0;
  auto alloc = [&](size_t n) { float* p = w + off; off += n; return p; };

  float* bufH   = alloc(25000000);  // h_gcn f32 -> enc16 f16[200000x128] -> xb/hlH/hrH
  float* bufG   = alloc(25000000);  // agg f32 -> hlT/hrT -> ssumH/aH/mH/denH/cntH
  float* dis    = alloc(N_LOW);
  float* logitT = alloc((size_t)E_LH * 4);
  float* mT     = alloc((size_t)N_HIGH * 4);
  float* denT   = alloc((size_t)N_HIGH * 4);
  float* cntT   = alloc(N_HIGH);
  float* ssumT  = alloc((size_t)N_HIGH * 100);
  float* x26    = alloc((size_t)N_HIGH * 26);
  float* stats  = alloc(256);
  float* a16f   = alloc(3200000);   // f16 activations [50000x128] (hh + MLP)
  float* a16gf  = alloc(128000);    // f16 GCN A [8000x32]
  float* w16af  = alloc(16384);     // f16 Wt scratch (<=640x32 / 128x128)
  float* w16bf  = alloc(16384);
  if (ws_size < off * sizeof(float)) return;  // ~264 MB arena requirement

  half_t* a16   = (half_t*)a16f;
  half_t* a16g  = (half_t*)a16gf;
  half_t* w16a  = (half_t*)w16af;
  half_t* w16b  = (half_t*)w16bf;
  half_t* enc16 = (half_t*)bufH;            // [200000,128] f16 after pixshuf

  // th-stage aliases in bufG
  float* hlT = bufG;                         // [200000,100]
  float* hrT = bufG + 20000000;              // [50000,100]
  // hh-stage aliases
  float* xb    = bufH;                       // [50000,128] f32 GAT output
  float* hlH   = bufH + 6400000;             // [50000,128]
  float* hrH   = bufH + 12800000;            // [50000,128]
  float* ssumH = bufG;                       // [50000,128]
  float* aH    = bufG + 6500000;             // [(E_HH+N_HIGH)*2]
  float* mH    = bufG + 8300000;             // [50000*2]
  float* denH  = bufG + 8500000;             // [50000*2]
  float* cntH  = bufG + 8700000;             // [50000]

  const int TB = 256;
  auto cdiv = [](int a, int b) { return (a + b - 1) / b; };
  auto fill = [&](float* p, float v, int n) {
    k_fill<<<dim3((unsigned)cdiv(n, TB)), dim3(TB), 0, stream>>>(p, v, n);
  };
  auto cvt_a = [&](const float* A, int lda, int M, int K, int Kp, half_t* o) {
    k_cvt_a16<<<dim3((unsigned)cdiv(M * Kp, TB)), dim3(TB), 0, stream>>>(A, lda, M, K, Kp, o);
  };
  auto cvt_wt = [&](const float* W_, int ldw, int K, int N, int Kp, int Np, half_t* o) {
    k_cvt_wt16<<<dim3((unsigned)cdiv(Np * Kp, TB)), dim3(TB), 0, stream>>>(W_, ldw, K, N, Kp, Np, o);
  };
  auto gemm16 = [&](const half_t* A, int Kp, const half_t* Wt, const float* bias,
                    float* C, int ldc, int M, int N, int Np, int relu) {
    dim3 g((unsigned)(M / 16), (unsigned)(Np / 16));
    if (Kp == 32)
      k_wmma_gemm16<32><<<g, 32, 0, stream>>>(A, Wt, bias, C, ldc, N, relu);
    else if (Kp == 64)
      k_wmma_gemm16<64><<<g, 32, 0, stream>>>(A, Wt, bias, C, ldc, N, relu);
    else
      k_wmma_gemm16<128><<<g, 32, 0, stream>>>(A, Wt, bias, C, ldc, N, relu);
  };

  // ======== Stage 1: GCN (5 variables) ========
  for (int v = 0; v < 5; ++v) {
    cvt_a(x_low + v * 25, 125, N_LOW, 25, 32, a16g);
    cvt_wt(gcn_w + (size_t)v * 25 * 625, 625, 25, 625, 32, 640, w16a);
    gemm16(a16g, 32, w16a, nullptr, bufH + (size_t)v * 625, 3125, N_LOW, 625, 640, 0);
  }
  fill(dis, 1.0f, N_LOW);
  k_deg<<<dim3((unsigned)cdiv(E_LL, TB)), dim3(TB), 0, stream>>>(ei_ll + E_LL, dis, E_LL);
  k_rsqrt_deg<<<dim3((unsigned)cdiv(N_LOW, TB)), dim3(TB), 0, stream>>>(dis, N_LOW);
  k_gcn_self<<<dim3((unsigned)cdiv(N_LOW * 3125, TB)), dim3(TB), 0, stream>>>(bufH, dis, gcn_b, bufG);
  k_gcn_edges<<<dim3((unsigned)E_LL), dim3(TB), 0, stream>>>(ei_ll, ei_ll + E_LL, dis, bufH, bufG);
  fill(bufH, 0.f, 12800000);   // zero f16 [200000,128] region (incl. K-pad cols)
  k_pixshuf16<<<dim3((unsigned)cdiv(N_LOW * 3125, TB)), dim3(TB), 0, stream>>>(bufG, enc16);

  // ======== Stage 2: bipartite GATv2 (H=4, oc=25, mean over heads) ========
  cvt_wt(th_wl, 100, 125, 100, 128, 112, w16a);
  gemm16(enc16, 128, w16a, th_bl, hlT, 100, N_L25, 100, 112, 0);
  k_hr_th<<<dim3((unsigned)cdiv(N_HIGH * 100, TB)), dim3(TB), 0, stream>>>(z_std, th_wr, th_br, hrT);
  fill(mT, -INFINITY, N_HIGH * 4);
  fill(denT, 0.f, N_HIGH * 4);
  fill(cntT, 0.f, N_HIGH);
  fill(ssumT, 0.f, N_HIGH * 100);
  {
    int tot = E_LH * 4;
    unsigned wb = (unsigned)cdiv(tot, 8);   // 8 waves per 256-thread block
    k_gat_logit_w<25><<<dim3(wb), dim3(TB), 0, stream>>>(
        src_l2h, dst_l2h, E_LH, 0, 4, hlT, hrT, th_att, logitT, mT, cntT);
    k_gat_exp<<<dim3((unsigned)cdiv(tot, TB)), dim3(TB), 0, stream>>>(
        dst_l2h, E_LH, 0, 4, mT, logitT, denT);
    k_gat_scatter_w<25><<<dim3(wb), dim3(TB), 0, stream>>>(
        src_l2h, dst_l2h, E_LH, 0, 4, hlT, logitT, denT, ssumT);
  }
  k_th_final<<<dim3((unsigned)cdiv(N_HIGH * 25, TB)), dim3(TB), 0, stream>>>(
      ssumT, cntT, th_bias, z_std, x26);

  // ======== Stage 3: hh GATv2 stack ========
  const int* srcH = ei_hh;
  const int* dstH = ei_hh + E_HH;

  auto run_bn16 = [&](const float* xin, int C, int Kp, const float* g, const float* b,
                      int relu) {
    k_bn_stats<<<dim3((unsigned)C), dim3(TB), 0, stream>>>(xin, N_HIGH, C, stats);
    k_bn_apply16<<<dim3((unsigned)cdiv(N_HIGH * Kp, TB)), dim3(TB), 0, stream>>>(
        xin, stats, g, b, N_HIGH, C, Kp, relu, a16);
  };

  auto run_gat_hh = [&](int K, int Kp,
                        const float* Wl, const float* bl,
                        const float* Wr, const float* br,
                        const float* att, const float* bias,
                        int H, int relu, float* xout) {
    int HC = H * 64;
    cvt_wt(Wl, HC, K, HC, Kp, HC, w16a);
    cvt_wt(Wr, HC, K, HC, Kp, HC, w16b);
    gemm16(a16, Kp, w16a, bl, hlH, HC, N_HIGH, HC, HC, 0);
    gemm16(a16, Kp, w16b, br, hrH, HC, N_HIGH, HC, HC, 0);
    fill(mH, -INFINITY, N_HIGH * H);
    fill(denH, 0.f, N_HIGH * H);
    fill(cntH, 0.f, N_HIGH);
    fill(ssumH, 0.f, N_HIGH * HC);
    int tot = (E_HH + N_HIGH) * H;
    unsigned wb = (unsigned)cdiv(tot, 8);
    k_gat_logit_w<64><<<dim3(wb), dim3(TB), 0, stream>>>(
        srcH, dstH, E_HH, N_HIGH, H, hlH, hrH, att, aH, mH, cntH);
    k_gat_exp<<<dim3((unsigned)cdiv(tot, TB)), dim3(TB), 0, stream>>>(
        dstH, E_HH, N_HIGH, H, mH, aH, denH);
    k_gat_scatter_w<64><<<dim3(wb), dim3(TB), 0, stream>>>(
        srcH, dstH, E_HH, N_HIGH, H, hlH, aH, denH, ssumH);
    k_hh_final<<<dim3((unsigned)cdiv(N_HIGH * HC, TB)), dim3(TB), 0, stream>>>(
        ssumH, cntH, bias, HC, relu, xout);
  };

  run_bn16(x26, 26, 32, bn0_g, bn0_b, 0);
  run_gat_hh(26, 32, g1_wl, g1_bl, g1_wr, g1_br, g1_att, g1_bias, 2, 0, xb);
  for (int i = 0; i < 3; ++i) {
    run_bn16(xb, 128, 128, bn_g + i * 128, bn_b + i * 128, 1);
    run_gat_hh(128, 128, gm_wl + (size_t)i * 16384, gm_bl + i * 128,
               gm_wr + (size_t)i * 16384, gm_br + i * 128,
               gm_att + i * 128, gm_bias + i * 128, 2, 0, xb);
  }
  run_bn16(xb, 128, 128, bn_g + 3 * 128, bn_b + 3 * 128, 1);
  run_gat_hh(128, 128, g5_wl, g5_bl, g5_wr, g5_br, g5_att, g5_bias, 1, 1, xb);

  // ======== Stage 4: MLP head (WMMA) ========
  cvt_a(xb, 64, N_HIGH, 64, 64, a16);
  cvt_wt(p_w1, 64, 64, 64, 64, 64, w16a);
  gemm16(a16, 64, w16a, p_b1, hlH, 64, N_HIGH, 64, 64, 1);

  cvt_a(hlH, 64, N_HIGH, 64, 64, a16);
  cvt_wt(p_w2, 32, 64, 32, 64, 32, w16a);
  gemm16(a16, 64, w16a, p_b2, hrH, 32, N_HIGH, 32, 32, 1);

  cvt_a(hrH, 32, N_HIGH, 32, 32, a16);
  cvt_wt(p_w3, 1, 32, 1, 32, 16, w16a);
  gemm16(a16, 32, w16a, p_b3, (float*)d_out, 1, N_HIGH, 1, 16, 0);
}
